// GRUBlock_56934086476260
// MI455X (gfx1250) — matrix operand order repaired and checked by reading, not compiled, not run on hardware
//
#include <hip/hip_runtime.h>
#include <cstdint>

// ---------------------------------------------------------------------------
// GRU block for MI455X (gfx1250, wave32).
//   B=64, S=2048, I=H=512.
// Phase 1: fused input projections as one bf16 WMMA GEMM (bandwidth-bound,
//          ~0.9 GB traffic -> ~40 us floor at 23.3 TB/s). Stage loads go
//          through GLOBAL_LOAD_ASYNC_TO_LDS (ASYNCcnt) when available.
// Phase 2: persistent recurrent kernel; recurrent weights VGPR-resident
//          (128 VGPRs/lane), h staged into LDS per step via the Tensor Data
//          Mover (tensor_load_to_lds, TENSORcnt, HW row padding) when
//          available; 4 independent batch-groups of 16 WGs synced per step
//          with cluster barrier (HW NOP unless cluster-launched) + software
//          counter barrier. All matmuls: v_wmma_f32_16x16x32_bf16 (f32 accum).
// ---------------------------------------------------------------------------

typedef __attribute__((ext_vector_type(16))) __bf16 v16bf;
typedef __attribute__((ext_vector_type(8)))  float  v8f;
typedef __attribute__((ext_vector_type(4)))  unsigned u32x4;
typedef __attribute__((ext_vector_type(8)))  int      i32x8;
typedef __attribute__((ext_vector_type(4)))  int      i32x4;

// async-to-LDS builtin takes vector-typed pointers (per round-2 diagnostic):
typedef int v4i __attribute__((vector_size(16)));
typedef v4i __attribute__((address_space(1)))* v4i_gp;   // global
typedef v4i __attribute__((address_space(3)))* v4i_lp;   // LDS
typedef __attribute__((address_space(3))) void* lds_ptr;

union ABf { v16bf v; unsigned u[8]; };

#define S_LEN 2048
#define BATCH 64
#define HID   512

#if __has_builtin(__builtin_amdgcn_global_load_async_to_lds_b128) && \
    __has_builtin(__builtin_amdgcn_s_wait_asynccnt)
#define HAS_ASYNC_LDS 1
#else
#define HAS_ASYNC_LDS 0
#endif

#if __has_builtin(__builtin_amdgcn_tensor_load_to_lds) && \
    __has_builtin(__builtin_amdgcn_s_wait_tensorcnt)
#define HAS_TDM 1
#else
#define HAS_TDM 0
#endif

// generic -> AS1/AS3 via integer round-trip (inttoptr; low 32 bits of a
// generic LDS address are the LDS byte offset per the aperture layout)
__device__ __forceinline__ v4i_gp to_g16(const void* p) {
  return (v4i_gp)(uintptr_t)p;
}
__device__ __forceinline__ v4i_lp to_l16(const void* p) {
  return (v4i_lp)(unsigned)(uintptr_t)p;
}
__device__ __forceinline__ unsigned lds_off_of(const void* p) {
  return (unsigned)(uintptr_t)p;   // low 32 bits of generic LDS address = offset
}

__device__ __forceinline__ float sigmoidf_(float x) { return 1.f / (1.f + __expf(-x)); }

// --------------------------- prep: weights -> bf16 --------------------------
__global__ void k_prep_w(const float* __restrict__ Wxz, const float* __restrict__ Wxr,
                         const float* __restrict__ Wxn, const float* __restrict__ Whz,
                         const float* __restrict__ Whr, const float* __restrict__ Whn,
                         __bf16* __restrict__ Wf, __bf16* __restrict__ Whf) {
  int idx = blockIdx.x * 256 + threadIdx.x;        // 0 .. 1536*512-1
  int n = idx >> 9, k = idx & 511;
  int gate = n >> 9, row = n & 511;
  const float* wx = gate == 0 ? Wxz : (gate == 1 ? Wxr : Wxn);
  const float* wh = gate == 0 ? Whz : (gate == 1 ? Whr : Whn);
  Wf[idx]  = (__bf16)wx[row * 512 + k];
  Whf[idx] = (__bf16)wh[row * 512 + k];
}

// --------------------------- prep: x -> bf16 --------------------------------
__global__ void k_conv_x(const float* __restrict__ x, __bf16* __restrict__ xb) {
  long long base = ((long long)blockIdx.x * 256 + threadIdx.x) * 8;
#pragma unroll
  for (int i = 0; i < 8; ++i) xb[base + i] = (__bf16)x[base + i];
}

// ------------------- prep: h double buffer + barrier ------------------------
// hbuf layout: [group(4)][parity(2)][m(16)][k(512)] bf16
__global__ void k_init(const float* __restrict__ h0, __bf16* __restrict__ hbuf,
                       unsigned* __restrict__ bar) {
  int idx = blockIdx.x * 256 + threadIdx.x;        // 0..32767
  int group = idx >> 13;
  int mk = idx & 8191;
  hbuf[group * 16384 + mk] = (__bf16)h0[group * 8192 + mk];   // parity 0
  if (idx < 4) bar[idx] = 0u;
}

// ------------------- phase 1: fused projection GEMM -------------------------
// out[m,n] = sum_k xb[m,k] * Wf[n,k];  m in [0,131072), n in [0,1536)
// stored time-major: proj[g][s][b][h] bf16
__global__ __launch_bounds__(256) void k_proj_gemm(const __bf16* __restrict__ xb,
                                                   const __bf16* __restrict__ Wf,
                                                   __bf16* __restrict__ proj) {
  __shared__ __bf16 As[2][128][80];   // 128 rows x 64 k (pad->80), double buffered
  __shared__ __bf16 Bs[2][64][80];

  const int tid  = threadIdx.x;
  const int lane = tid & 31, wid = tid >> 5;
  const int waveM = wid & 3, waveN = wid >> 2;        // 4 (M) x 2 (N) waves
  const long long M0 = (long long)blockIdx.y * 128;
  const int N0 = blockIdx.x * 64;

  v8f acc[2][2] = {{{}, {}}, {{}, {}}};

  auto load_stage = [&](int s, int buf) {
    const int ks = s * 64;
#pragma unroll
    for (int i = 0; i < 4; ++i) {                     // A: 128x64 bf16, 16B chunks
      int c = tid + i * 256;
      int row = c >> 3, col = (c & 7) * 8;
      const __bf16* src = xb + (M0 + row) * 512 + ks + col;
#if HAS_ASYNC_LDS
      __builtin_amdgcn_global_load_async_to_lds_b128(to_g16(src),
                                                     to_l16(&As[buf][row][col]), 0, 0);
#else
      float4 d = *reinterpret_cast<const float4*>(src);
      *reinterpret_cast<float4*>(&As[buf][row][col]) = d;
#endif
    }
#pragma unroll
    for (int i = 0; i < 2; ++i) {                     // B: 64x64 bf16
      int c = tid + i * 256;
      int row = c >> 3, col = (c & 7) * 8;
      const __bf16* src = Wf + (long long)(N0 + row) * 512 + ks + col;
#if HAS_ASYNC_LDS
      __builtin_amdgcn_global_load_async_to_lds_b128(to_g16(src),
                                                     to_l16(&Bs[buf][row][col]), 0, 0);
#else
      float4 d = *reinterpret_cast<const float4*>(src);
      *reinterpret_cast<float4*>(&Bs[buf][row][col]) = d;
#endif
    }
  };

  // A fragment: 16x32 bf16 per ISA VGPR layout
  auto frag_a = [&](int buf, int mrow, int k) -> v16bf {
    ABf f;
    const int m = mrow + (lane & 15);
    const int half = lane >> 4;
    const unsigned* p = reinterpret_cast<const unsigned*>(&As[buf][m][0]);
#pragma unroll
    for (int v = 0; v < 8; ++v) {
      int kk = k + ((v < 4) ? (2 * v + 8 * half) : (16 + 2 * (v - 4) + 8 * half));
      f.u[v] = p[kk >> 1];
    }
    return f.v;
  };
  // B fragment: 32x16 bf16
  auto frag_b = [&](int buf, int nrow, int k) -> v16bf {
    ABf f;
    const int n = nrow + (lane & 15);
    const int half = lane >> 4;
    const unsigned* p = reinterpret_cast<const unsigned*>(&Bs[buf][n][0]);
#pragma unroll
    for (int v = 0; v < 8; ++v) {
      int kk = k + ((v < 4) ? (2 * v + 8 * half) : (16 + 2 * (v - 4) + 8 * half));
      f.u[v] = p[kk >> 1];
    }
    return f.v;
  };

  load_stage(0, 0);
  for (int s = 0; s < 8; ++s) {
    if (s + 1 < 8) {
      if (s + 2 < 8)  // near-cache prefetch of the stage after next
        __builtin_prefetch(xb + (M0 + (tid >> 1)) * 512 + (s + 2) * 64, 0, 3);
      load_stage(s + 1, (s + 1) & 1);
#if HAS_ASYNC_LDS
      __builtin_amdgcn_s_wait_asynccnt(6);   // 6 newest outstanding -> stage s resident
#endif
    } else {
#if HAS_ASYNC_LDS
      __builtin_amdgcn_s_wait_asynccnt(0);
#endif
    }
    __syncthreads();
#pragma unroll
    for (int kk = 0; kk < 64; kk += 32) {
      v16bf a0 = frag_a(s & 1, waveM * 32 + 0,  kk);
      v16bf a1 = frag_a(s & 1, waveM * 32 + 16, kk);
      v16bf b0 = frag_b(s & 1, waveN * 32 + 0,  kk);
      v16bf b1 = frag_b(s & 1, waveN * 32 + 16, kk);
      acc[0][0] = __builtin_amdgcn_wmma_f32_16x16x32_bf16(false, a0, false, b0, (short)0, acc[0][0], false, false);
      acc[0][1] = __builtin_amdgcn_wmma_f32_16x16x32_bf16(false, a0, false, b1, (short)0, acc[0][1], false, false);
      acc[1][0] = __builtin_amdgcn_wmma_f32_16x16x32_bf16(false, a1, false, b0, (short)0, acc[1][0], false, false);
      acc[1][1] = __builtin_amdgcn_wmma_f32_16x16x32_bf16(false, a1, false, b1, (short)0, acc[1][1], false, false);
    }
    __syncthreads();   // compute done before next stage overwrites the other buffer
  }

  // D layout: VGPR v, lane L -> m = v + 8*(L/16), n = L%16
#pragma unroll
  for (int mi = 0; mi < 2; ++mi)
#pragma unroll
    for (int ni = 0; ni < 2; ++ni) {
      int n = N0 + waveN * 32 + ni * 16 + (lane & 15);
      int g = n >> 9, h = n & 511;
#pragma unroll
      for (int v = 0; v < 8; ++v) {
        long long m = M0 + waveM * 32 + mi * 16 + v + 8 * (lane >> 4);
        int b = (int)(m >> 11), sx = (int)(m & 2047);
        proj[(((long long)g * S_LEN + sx) * BATCH + b) * HID + h] = (__bf16)acc[mi][ni][v];
      }
    }
}

// ------------------- phase 2: persistent recurrent scan ---------------------
__global__ __launch_bounds__(192) void k_scan(const __bf16* __restrict__ Whf,
                                              const __bf16* __restrict__ proj,
                                              const float* __restrict__ b_hz,
                                              const float* __restrict__ b_hr,
                                              const float* __restrict__ b_hn,
                                              __bf16* __restrict__ hbuf,
                                              unsigned* __restrict__ bar,
                                              float* __restrict__ out,
                                              float* __restrict__ hlast) {
  __shared__ __bf16 hl[16][528];        // full h for this batch group, 528 = 512 + TDM pad
  __shared__ float  gres[3][16][32];    // z/r/n matmul results for this slice

  const int tid = threadIdx.x;
  const int lane = tid & 31, wid = tid >> 5;
  const int group = blockIdx.x >> 4;
  const int slice = blockIdx.x & 15;
  const int nbase = slice << 5;
  const int gate = wid >> 1, ntile = wid & 1;
  const int half = lane >> 4, lm = lane & 15;

  // ---- pin recurrent weight fragments in VGPRs: 16 k-steps x 8 dwords/lane
  ABf bw[16];
  {
    const long long n_g = gate * 512 + nbase + ntile * 16 + lm;
    const unsigned* wp = reinterpret_cast<const unsigned*>(Whf) + n_g * 256;
#pragma unroll
    for (int ks = 0; ks < 16; ++ks) {
      const int kb = ks * 32 + (half << 4);
#pragma unroll
      for (int v = 0; v < 8; ++v) bw[ks].u[v] = wp[(ks * 32 + ((v < 4) ? (2 * v + 8 * half) : (16 + 2 * (v - 4) + 8 * half))) >> 1];
    }
  }

  unsigned* cnt = bar + group;
  __bf16* hb = hbuf + group * 16384;
  const long long PSTRIDE = (long long)S_LEN * BATCH * HID;
#if HAS_TDM
  const unsigned hl_off = lds_off_of(&hl[0][0]);
#endif

  for (int t = 0; t < S_LEN; ++t) {
    // ---- stage h[16][512] bf16 into LDS (row pitch 528)
    const __bf16* hsrc = hb + (t & 1) * 8192;
#if HAS_TDM
    if (wid == 0) {
      // TDM D#: 2D tile 512x16 of 2-byte elems, HW pad 8 dwords per 256 dwords
      unsigned long long ga = (unsigned long long)(uintptr_t)hsrc;
      u32x4 g0;
      g0.x = 1u;                                            // count = 1
      g0.y = hl_off;                                        // lds_addr
      g0.z = (unsigned)ga;                                  // global_addr lo
      g0.w = (unsigned)((ga >> 32) & 0x01FFFFFFull) | (2u << 30);  // addr hi | type=2
      i32x8 g1;
      g1[0] = (int)0x0FD10000u;  // data_size=2B, pad_en, interval=256dw, amount=8dw
      g1[1] = (int)(512u << 16); // tensor_dim0 = 512
      g1[2] = (int)(16u << 16);  // tensor_dim1 = 16
      g1[3] = (int)(512u << 16); // tile_dim0 = 512
      g1[4] = 16;                // tile_dim1 = 16
      g1[5] = 512;               // tensor_dim0_stride = 512
      g1[6] = 0; g1[7] = 0;
      i32x4 z4 = {0, 0, 0, 0};
#if __clang_major__ >= 23
      i32x8 z8 = {0, 0, 0, 0, 0, 0, 0, 0};
      __builtin_amdgcn_tensor_load_to_lds(g0, g1, z4, z4, z8, 0);
#else
      __builtin_amdgcn_tensor_load_to_lds(g0, g1, z4, z4, 0);
#endif
      __builtin_amdgcn_s_wait_tensorcnt(0);
    }
#else
    for (int c = tid; c < 1024; c += 192) {
      int row = c >> 6, col = (c & 63) * 8;
      float4 d = *reinterpret_cast<const float4*>(hsrc + row * 512 + col);
      *reinterpret_cast<float4*>(&hl[row][col]) = d;
    }
#endif
    __syncthreads();

    // ---- h @ W_g^T for this wave's 16x16 tile, K = 512 (16 WMMAs)
    {
      v8f acc = {};
      const unsigned* hp = reinterpret_cast<const unsigned*>(&hl[0][0]);
#pragma unroll
      for (int ks = 0; ks < 16; ++ks) {
        ABf a;
#pragma unroll
        for (int v = 0; v < 8; ++v) {
          int k = ks * 32 + ((v < 4) ? (2 * v + 8 * half) : (16 + 2 * (v - 4) + 8 * half));
          a.u[v] = hp[(lm * 528 + k) >> 1];
        }
        acc = __builtin_amdgcn_wmma_f32_16x16x32_bf16(false, a.v, false, bw[ks].v, (short)0, acc, false, false);
      }
#pragma unroll
      for (int v = 0; v < 8; ++v)
        gres[gate][v + 8 * half][ntile * 16 + lm] = acc[v];
    }
    __syncthreads();

    // ---- gate math (fp32) for this [16 x 32] slice
    for (int idx = tid; idx < 512; idx += 192) {
      int m = idx >> 5, j = idx & 31;
      int h = nbase + j;
      int b = group * 16 + m;
      long long pbase = ((long long)t * BATCH + b) * HID + h;
      float xz = (float)proj[pbase];
      float xr = (float)proj[pbase + PSTRIDE];
      float xn = (float)proj[pbase + 2 * PSTRIDE];
      float az = gres[0][m][j] + b_hz[h];
      float ar = gres[1][m][j] + b_hr[h];
      float an = gres[2][m][j] + b_hn[h];
      float z = sigmoidf_(xz + az);
      float r = sigmoidf_(xr + ar);
      float n = tanhf(xn + r * an);
      float hprev = (float)hl[m][h];
      float hnew = (1.f - z) * n + z * hprev;
      out[((long long)b * S_LEN + t) * HID + h] = hnew;
      hb[((t + 1) & 1) * 8192 + m * 512 + h] = (__bf16)hnew;
      if (t == S_LEN - 1) hlast[b * HID + h] = hnew;
    }

    // ---- per-step sync across the 16 WGs of this batch group
    __threadfence();
    __syncthreads();
    if (tid == 0) {
      __builtin_amdgcn_s_cluster_barrier();   // NOP unless cluster-launched
      __hip_atomic_fetch_add(cnt, 1u, __ATOMIC_RELEASE, __HIP_MEMORY_SCOPE_AGENT);
      unsigned target = 16u * (unsigned)(t + 1);
      while (__hip_atomic_load(cnt, __ATOMIC_ACQUIRE, __HIP_MEMORY_SCOPE_AGENT) < target)
        __builtin_amdgcn_s_sleep(8);
    }
    __syncthreads();
  }
}

// ---------------------------------------------------------------------------
extern "C" void kernel_launch(void* const* d_in, const int* in_sizes, int n_in,
                              void* d_out, int out_size, void* d_ws, size_t ws_size,
                              hipStream_t stream) {
  (void)in_sizes; (void)n_in; (void)out_size; (void)ws_size;
  const float* x    = (const float*)d_in[0];
  const float* h0   = (const float*)d_in[1];
  const float* W_xz = (const float*)d_in[2];
  const float* W_hz = (const float*)d_in[3];
  const float* b_hz = (const float*)d_in[4];
  const float* W_xr = (const float*)d_in[5];
  const float* W_hr = (const float*)d_in[6];
  const float* b_hr = (const float*)d_in[7];
  const float* W_xn = (const float*)d_in[8];
  const float* W_hn = (const float*)d_in[9];
  const float* b_hn = (const float*)d_in[10];

  // workspace layout (bytes):
  //   [0, 1.5M)       Wf   bf16 [1536][512]
  //   [1.5M, 3M)      Whf  bf16 [1536][512]
  //   [3M, +128K)     hbuf bf16 [4][2][16][512]
  //   [+128K]         bar  u32  [4]
  //   [4M, +128M)     xb   bf16 [131072][512]
  //   [+128M, +384M)  proj bf16 [3][2048][64][512]
  char* ws = (char*)d_ws;
  __bf16*   Wf   = (__bf16*)(ws);
  __bf16*   Whf  = (__bf16*)(ws + 1536ll * 512 * 2);
  __bf16*   hbuf = (__bf16*)(ws + 2 * 1536ll * 512 * 2);
  unsigned* bar  = (unsigned*)(ws + 2 * 1536ll * 512 * 2 + 4ll * 2 * 16 * 512 * 2);
  __bf16*   xb   = (__bf16*)(ws + (4ll << 20));
  __bf16*   proj = (__bf16*)(ws + (4ll << 20) + 131072ll * 512 * 2);

  float* out   = (float*)d_out;
  float* hlast = out + (long long)BATCH * S_LEN * HID;

  k_prep_w<<<3072, 256, 0, stream>>>(W_xz, W_xr, W_xn, W_hz, W_hr, W_hn, Wf, Whf);
  k_conv_x<<<32768, 256, 0, stream>>>(x, xb);
  k_init<<<128, 256, 0, stream>>>(h0, hbuf, bar);
  dim3 g1(24, 1024);   // N blocks (1536/64) x M blocks (131072/128)
  k_proj_gemm<<<g1, 256, 0, stream>>>(xb, Wf, proj);
  k_scan<<<64, 192, 0, stream>>>(Whf, proj, b_hz, b_hr, b_hn, hbuf, bar, out, hlast);
}
